// ResidualVectorQuantizer_66116726555423
// MI455X (gfx1250) — matrix operand order repaired and checked by reading, not compiled, not run on hardware
//
#include <hip/hip_runtime.h>
#include <hip/hip_bf16.h>

typedef __attribute__((ext_vector_type(16))) __bf16 v16bf;
typedef __attribute__((ext_vector_type(8)))  float  v8f;

#define B_    16
#define T_    4096
#define NCB   8
#define K_    1024
#define DCB   64
#define D_    512
#define WAVES 8
#define TOK_PER_WG (WAVES * 16)   // 128 tokens per workgroup
#define NTILE (K_ / 16)           // 64 N-tiles of 16 codes

// LDS: codebook hi/lo bf16 (2*128KB) + 0.5*||w||^2 fp32 (4KB) + per-wave idx (512B)
#define SMEM_BYTES (2 * K_ * DCB * 2 + K_ * 4 + WAVES * 16 * 4)

static __device__ __forceinline__ __bf16 f2bf(float f) {
  union { float f; unsigned u; } in; in.f = f;
  unsigned r = (in.u + 0x7FFFu + ((in.u >> 16) & 1u)) >> 16;  // RNE
  union { unsigned short s; __bf16 h; } out; out.s = (unsigned short)r;
  return out.h;
}
static __device__ __forceinline__ float bf2f(__bf16 h) {
  union { unsigned short s; __bf16 h; } in; in.h = h;
  union { unsigned u; float f; } out; out.u = ((unsigned)in.s) << 16;
  return out.f;
}
static __device__ __forceinline__ v8f wmma_bf16(v16bf a, v16bf b, v8f c) {
  // (neg_a, A, neg_b, B, c_mod, C, reuse_a, reuse_b)
  return __builtin_amdgcn_wmma_f32_16x16x32_bf16(false, a, false, b, (short)0, c,
                                                 false, false);
}

__global__ __launch_bounds__(256, 1)
void ResidualVectorQuantizer_66116726555423_kernel(const float* __restrict__ z,
                                                   const float* __restrict__ cbs,
                                                   float* __restrict__ out) {
  extern __shared__ char smem[];
  __bf16* whi   = (__bf16*)smem;               // [K_][DCB] bf16 hi
  __bf16* wlo   = whi + K_ * DCB;              // [K_][DCB] bf16 lo
  float*  wsq   = (float*)(wlo + K_ * DCB);    // [K_] 0.5*||w||^2 fp32
  int*    idx_s = (int*)(wsq + K_);            // [WAVES*16]

  const int tid  = threadIdx.x;
  const int wave = tid >> 5;
  const int lane = tid & 31;
  const int lh   = lane >> 4;      // lane half (A/B K-group selector)
  const int m    = lane & 15;      // token row within wave (A layout: M = lane&15)
  const int gt   = blockIdx.x * TOK_PER_WG + wave * 16 + m;  // global token

  float* zq_out  = out;                        // [B*T*D] fp32
  float* idx_out = out + (size_t)B_ * T_ * D_; // [B*NCB*T] indices (as fp32 values)

  // residual for this lane's token: 32 of its 64 dims, in A-fragment halfword order.
  // element e = 16*chunk + hw  <->  k = 32*chunk + 16*(hw/8) + 8*lh + (hw%8)
  float res[32];
#pragma unroll
  for (int e = 0; e < 32; ++e) res[e] = 0.f;

  for (int cb = 0; cb < NCB; ++cb) {
    __syncthreads();  // previous stage's LDS reads done before restaging

    // ---- stage codebook cb into LDS as bf16 hi/lo (shared by all 8 waves) ----
    const float*  src  = cbs + (size_t)cb * K_ * DCB;
    const float4* src4 = (const float4*)src;
    for (int i = tid; i < K_ * DCB / 4; i += 256) {
      float4 v = src4[i];
      float f[4] = {v.x, v.y, v.z, v.w};
#pragma unroll
      for (int j = 0; j < 4; ++j) {
        __bf16 h = f2bf(f[j]);
        int d = i * 4 + j;
        int p = (d & ~24) | ((d & 8) << 1) | ((d & 16) >> 1);
        whi[p] = h;
        wlo[p] = f2bf(f[j] - bf2f(h));
      }
    }
    // 0.5*||w||^2 per code, fp32 (L2-hot re-read)
    for (int n = tid; n < K_; n += 256) {
      const float4* wr = (const float4*)(src + n * DCB);
      float s = 0.f;
#pragma unroll
      for (int q = 0; q < DCB / 4; ++q) {
        float4 v = wr[q];
        s += v.x * v.x + v.y * v.y + v.z * v.z + v.w * v.w;
      }
      wsq[n] = 0.5f * s;
    }
    __syncthreads();

    // ---- residual += z slice for this codebook ----
    const float* zbase = z + (size_t)gt * D_ + cb * DCB;
#pragma unroll
    for (int r = 0; r < 4; ++r) {
      int ks = 32 * (r >> 1) + 16 * (r & 1) + 8 * lh;
      const float4* zp = (const float4*)(zbase + ks);
      float4 v0 = zp[0], v1 = zp[1];
      res[8*r+0] += v0.x; res[8*r+1] += v0.y; res[8*r+2] += v0.z; res[8*r+3] += v0.w;
      res[8*r+4] += v1.x; res[8*r+5] += v1.y; res[8*r+6] += v1.z; res[8*r+7] += v1.w;
    }

    // ---- A fragments of (-res), bf16 hi/lo split (fp32-accurate products) ----
    v16bf ahi[2], alo[2];
#pragma unroll
    for (int c = 0; c < 2; ++c) {
#pragma unroll
      for (int hw = 0; hw < 16; ++hw) {
        float neg = -res[16 * c + hw];
        __bf16 h = f2bf(neg);
        ahi[c][hw] = h;
        alo[c][hw] = f2bf(neg - bf2f(h));
      }
    }

    // ---- sweep 64 N-tiles; score = 0.5||w||^2 - res.w ; running argmin ----
    float bestS[8];
    int   bestN[8];
#pragma unroll
    for (int r = 0; r < 8; ++r) { bestS[r] = 3.4e38f; bestN[r] = 0; }

    // software pipeline: prefetch tile fragments one iteration ahead so the
    // ds_load latency overlaps the WMMA chains of the current tile.
    const __bf16* bhp0 = whi + m * DCB + 16 * lh;
    const __bf16* blp0 = wlo + m * DCB + 16 * lh;
    v16bf bh0 = *(const v16bf*)(bhp0);
    v16bf bh1 = *(const v16bf*)(bhp0 + 32);
    v16bf bl0 = *(const v16bf*)(blp0);
    v16bf bl1 = *(const v16bf*)(blp0 + 32);
    float iv  = wsq[m];

    for (int tile = 0; tile < NTILE; ++tile) {
      const int n  = tile * 16 + m;              // this lane's B/C column
      const int nn = (((tile + 1) & (NTILE - 1)) * 16) + m;  // wrap: always valid
      const __bf16* nbhp = whi + nn * DCB + 16 * lh;
      const __bf16* nblp = wlo + nn * DCB + 16 * lh;
      v16bf nbh0 = *(const v16bf*)(nbhp);
      v16bf nbh1 = *(const v16bf*)(nbhp + 32);
      v16bf nbl0 = *(const v16bf*)(nblp);
      v16bf nbl1 = *(const v16bf*)(nblp + 32);
      float niv  = wsq[nn];

      // two independent accumulator chains (halve the C->C dependency depth)
      v8f acc0 = {iv, iv, iv, iv, iv, iv, iv, iv};
      v8f acc1 = {0.f, 0.f, 0.f, 0.f, 0.f, 0.f, 0.f, 0.f};
      acc0 = wmma_bf16(ahi[0], bh0, acc0);
      acc1 = wmma_bf16(ahi[1], bh1, acc1);
      acc0 = wmma_bf16(alo[0], bh0, acc0);
      acc1 = wmma_bf16(alo[1], bh1, acc1);
      acc0 = wmma_bf16(ahi[0], bl0, acc0);
      acc1 = wmma_bf16(ahi[1], bl1, acc1);

#pragma unroll
      for (int r = 0; r < 8; ++r) {
        float s = acc0[r] + acc1[r];  // C layout: VGPR r -> M = r + 8*lh ; lane -> N
        bool t = s < bestS[r];
        bestS[r] = t ? s : bestS[r];
        bestN[r] = t ? n : bestN[r];
      }

      bh0 = nbh0; bh1 = nbh1; bl0 = nbl0; bl1 = nbl1; iv = niv;
    }

    // cross-lane argmin within each 16-lane half (xor masks stay in-half)
#pragma unroll
    for (int sh = 1; sh <= 8; sh <<= 1) {
#pragma unroll
      for (int r = 0; r < 8; ++r) {
        float os = __shfl_xor(bestS[r], sh, 32);
        int   on = __shfl_xor(bestN[r], sh, 32);
        bool  t  = (os < bestS[r]) | ((os == bestS[r]) & (on < bestN[r]));
        bestS[r] = t ? os : bestS[r];
        bestN[r] = t ? on : bestN[r];
      }
    }

    if (m == 0) {  // lane 0 holds tokens 0..7, lane 16 holds tokens 8..15
#pragma unroll
      for (int r = 0; r < 8; ++r) {
        int tok = r + 8 * lh;
        idx_s[wave * 16 + tok] = bestN[r];
        int g  = blockIdx.x * TOK_PER_WG + wave * 16 + tok;
        int bb = g >> 12, tt = g & (T_ - 1);
        idx_out[(size_t)bb * NCB * T_ + (size_t)cb * T_ + tt] = (float)bestN[r];
      }
    }
    __syncthreads();

    // ---- gather winning code in fp32 (exact), emit z_q, update residual ----
    const int myIdx = idx_s[wave * 16 + m];
    const float* wrow = cbs + (size_t)cb * K_ * DCB + (size_t)myIdx * DCB;
    float* qout = zq_out + (size_t)gt * D_ + cb * DCB;
#pragma unroll
    for (int r = 0; r < 4; ++r) {
      int ks = 32 * (r >> 1) + 16 * (r & 1) + 8 * lh;
      const float4* wp = (const float4*)(wrow + ks);
      float4 v0 = wp[0], v1 = wp[1];
      res[8*r+0] -= v0.x; res[8*r+1] -= v0.y; res[8*r+2] -= v0.z; res[8*r+3] -= v0.w;
      res[8*r+4] -= v1.x; res[8*r+5] -= v1.y; res[8*r+6] -= v1.z; res[8*r+7] -= v1.w;
      *(float4*)(qout + ks)     = v0;
      *(float4*)(qout + ks + 4) = v1;
    }
  }
}

extern "C" void kernel_launch(void* const* d_in, const int* in_sizes, int n_in,
                              void* d_out, int out_size, void* d_ws, size_t ws_size,
                              hipStream_t stream) {
  (void)in_sizes; (void)n_in; (void)out_size; (void)d_ws; (void)ws_size;
  const float* z   = (const float*)d_in[0];
  const float* cbs = (const float*)d_in[1];
  float* out = (float*)d_out;

  hipFuncSetAttribute(
      reinterpret_cast<const void*>(ResidualVectorQuantizer_66116726555423_kernel),
      hipFuncAttributeMaxDynamicSharedMemorySize, SMEM_BYTES);

  dim3 grid((B_ * T_) / TOK_PER_WG);  // 512 workgroups
  dim3 block(256);                    // 8 waves (wave32)
  hipLaunchKernelGGL(ResidualVectorQuantizer_66116726555423_kernel,
                     grid, block, SMEM_BYTES, stream, z, cbs, out);
}
